// GlobalContextInjection_35656818491963
// MI455X (gfx1250) — matrix operand order repaired and checked by reading, not compiled, not run on hardware
//
#include <hip/hip_runtime.h>
#include <math.h>

// Problem constants (from reference): N=200000 nodes, H=256, H/2=128, G=512 graphs.
#define HDIM 256
#define HHALF 128
#define GSEG 512

typedef __attribute__((ext_vector_type(16))) __bf16 v16bf;
typedef __attribute__((ext_vector_type(8)))  float  v8f;
typedef __attribute__((ext_vector_type(4)))  float  f4;

// -------- helpers --------------------------------------------------------

__device__ __forceinline__ int lower_bound_i32(const int* __restrict__ b, int n, int key) {
    int lo = 0, hi = n;
    while (lo < hi) {
        int mid = (lo + hi) >> 1;
        if (b[mid] < key) lo = mid + 1; else hi = mid;
    }
    return lo;
}

// Build a 16-bit A-matrix fragment (16x32 bf16) for one lane from a row-major
// f32 row pointer.  Lanes 0-15 hold K=kt*32+0..15, lanes 16-31 hold
// K=kt*32+16..31 (contiguous, matching packed B); row = lane & 15.
__device__ __forceinline__ v16bf load_a_bf16(const float* __restrict__ rowptr,
                                             int kt, int khalf) {
    const f4* p0 = (const f4*)(rowptr + kt * 32 + khalf * 16);
    const f4* p1 = (const f4*)(rowptr + kt * 32 + khalf * 16 + 8);
    f4 f0 = p0[0], f1 = p0[1], f2 = p1[0], f3 = p1[1];
    v16bf a;
    a[0]  = (__bf16)f0.x; a[1]  = (__bf16)f0.y; a[2]  = (__bf16)f0.z; a[3]  = (__bf16)f0.w;
    a[4]  = (__bf16)f1.x; a[5]  = (__bf16)f1.y; a[6]  = (__bf16)f1.z; a[7]  = (__bf16)f1.w;
    a[8]  = (__bf16)f2.x; a[9]  = (__bf16)f2.y; a[10] = (__bf16)f2.z; a[11] = (__bf16)f2.w;
    a[12] = (__bf16)f3.x; a[13] = (__bf16)f3.y; a[14] = (__bf16)f3.z; a[15] = (__bf16)f3.w;
    return a;
}

// -------- kernel 1: gating MLP score  s = tanh(x@W1+b1)@W2 + b2 ----------
// Block = 256 threads = 8 waves; each wave computes a 16-node strip.
// W1 is pre-packed into LDS in WMMA B-register layout:
//   ldsB[kt][ntl][lane][j] = bf16( W1[(kt*32 + (lane>>4)*16 + j)*128 + ntl*16 + (lane&15)] )
// (B 32x16 layout: lane = column N, lanes 0-15 K=0..15 contiguous, lanes 16-31 K=16..31.)
__global__ __launch_bounds__(256)
void gci_score_kernel(const float* __restrict__ x,
                      const float* __restrict__ W1,
                      const float* __restrict__ b1,
                      const float* __restrict__ W2,
                      const float* __restrict__ b2,
                      float* __restrict__ s, int n) {
    __shared__ __bf16 ldsB[8 * 8 * 32 * 16];   // 64 KB packed W1
    __shared__ float  ldsB1[HHALF];
    __shared__ float  ldsW2[HHALF];

    const int tid = threadIdx.x;
    for (int idx = tid; idx < 8 * 8 * 32 * 16; idx += 256) {
        int j    = idx & 15;
        int lane = (idx >> 4) & 31;
        int ntl  = (idx >> 9) & 7;
        int kt   = idx >> 12;
        int k    = kt * 32 + (lane >> 4) * 16 + j;
        int c    = ntl * 16 + (lane & 15);
        ldsB[idx] = (__bf16)W1[k * HHALF + c];
    }
    if (tid < HHALF) { ldsB1[tid] = b1[tid]; ldsW2[tid] = W2[tid]; }
    __syncthreads();

    const int lane  = tid & 31;
    const int wave  = tid >> 5;
    const int base  = blockIdx.x * 128 + wave * 16;
    const int khalf = lane >> 4;
    int row = base + (lane & 15);
    if (row >= n) row = n - 1;                 // clamp: keeps EXEC full for WMMA
    const float* xr = x + (long)row * HDIM;

    v8f acc[8] = {};
    #pragma unroll
    for (int kt = 0; kt < 8; ++kt) {
        v16bf a = load_a_bf16(xr, kt, khalf);
        #pragma unroll
        for (int ntl = 0; ntl < 8; ++ntl) {
            v16bf bm = *(const v16bf*)&ldsB[(((kt << 3) | ntl) << 9) + (lane << 4)];
            acc[ntl] = __builtin_amdgcn_wmma_f32_16x16x32_bf16(
                false, a, false, bm, (short)0, acc[ntl], false, false);
        }
    }

    // h = tanh(acc + b1); partial s = h . W2   (C layout: VGPR r -> M=r+8*khalf,
    // N = ntl*16 + (lane&15))
    float part[8] = {0.f, 0.f, 0.f, 0.f, 0.f, 0.f, 0.f, 0.f};
    #pragma unroll
    for (int ntl = 0; ntl < 8; ++ntl) {
        int   c  = ntl * 16 + (lane & 15);
        float bb = ldsB1[c];
        float ww = ldsW2[c];
        #pragma unroll
        for (int r = 0; r < 8; ++r)
            part[r] += tanhf(acc[ntl][r] + bb) * ww;
    }
    // reduce over the 16 lanes of each half-wave (wave32 butterfly, bits 1..8)
    #pragma unroll
    for (int m = 1; m < 16; m <<= 1) {
        #pragma unroll
        for (int r = 0; r < 8; ++r)
            part[r] += __shfl_xor(part[r], m, 32);
    }
    if ((lane & 15) == 0) {
        float b2s   = b2[0];
        int   rbase = base + khalf * 8;
        #pragma unroll
        for (int r = 0; r < 8; ++r) {
            int rr = rbase + r;
            if (rr < n) s[rr] = part[r] + b2s;
        }
    }
}

// -------- kernel 2: segment softmax weights ------------------------------
__global__ __launch_bounds__(256)
void gci_segsoft_kernel(const int* __restrict__ batch,
                        const float* __restrict__ s,
                        float* __restrict__ wnode, int n) {
    __shared__ float red[256];
    const int g = blockIdx.x, tid = threadIdx.x;
    const int start = lower_bound_i32(batch, n, g);
    const int end   = lower_bound_i32(batch, n, g + 1);

    float mx = -3.402823466e38f;
    for (int i = start + tid; i < end; i += 256) mx = fmaxf(mx, s[i]);
    red[tid] = mx; __syncthreads();
    for (int st = 128; st > 0; st >>= 1) {
        if (tid < st) red[tid] = fmaxf(red[tid], red[tid + st]);
        __syncthreads();
    }
    mx = red[0]; __syncthreads();

    float sum = 0.f;
    for (int i = start + tid; i < end; i += 256) sum += expf(s[i] - mx);
    red[tid] = sum; __syncthreads();
    for (int st = 128; st > 0; st >>= 1) {
        if (tid < st) red[tid] += red[tid + st];
        __syncthreads();
    }
    float inv = (red[0] > 0.f) ? (1.0f / red[0]) : 0.f;
    for (int i = start + tid; i < end; i += 256) wnode[i] = expf(s[i] - mx) * inv;
}

// -------- kernel 3: weighted segment pooling -----------------------------
// Block per graph; thread t owns column t; rows of x read coalesced (last use
// of x -> nontemporal).
__global__ __launch_bounds__(256)
void gci_pool_kernel(const float* __restrict__ x,
                     const int* __restrict__ batch,
                     const float* __restrict__ wnode,
                     float* __restrict__ gemb, int n) {
    const int g = blockIdx.x, tid = threadIdx.x;
    const int start = lower_bound_i32(batch, n, g);
    const int end   = lower_bound_i32(batch, n, g + 1);
    float acc = 0.f;
    for (int i = start; i < end; ++i) {
        float w = wnode[i];
        acc += __builtin_nontemporal_load(x + (long)i * HDIM + tid) * w;
    }
    gemb[g * HDIM + tid] = acc;
}

// -------- kernel 4: context = graph_emb @ Wp + bp (WMMA) -----------------
// 64 blocks; block b: M-tile mt=b>>1, N-tiles [(b&1)*8, (b&1)*8+8); 8 waves,
// wave w handles one 16x16 output tile, K=256 in 8 bf16 WMMA steps.
__global__ __launch_bounds__(256)
void gci_proj_kernel(const float* __restrict__ gemb,
                     const float* __restrict__ Wp,
                     const float* __restrict__ bp,
                     float* __restrict__ ctx) {
    __shared__ __bf16 ldsB[8 * 8 * 32 * 16];   // 64 KB packed Wp slice
    const int tid = threadIdx.x;
    const int mt  = blockIdx.x >> 1;
    const int nb  = (blockIdx.x & 1) * 8;

    for (int idx = tid; idx < 8 * 8 * 32 * 16; idx += 256) {
        int j    = idx & 15;
        int lane = (idx >> 4) & 31;
        int ntl  = (idx >> 9) & 7;
        int kt   = idx >> 12;
        int k    = kt * 32 + (lane >> 4) * 16 + j;
        int c    = (nb + ntl) * 16 + (lane & 15);
        ldsB[idx] = (__bf16)Wp[k * HDIM + c];
    }
    __syncthreads();

    const int lane  = tid & 31;
    const int wave  = tid >> 5;
    const int ntl   = nb + wave;
    const int khalf = lane >> 4;
    const int row   = mt * 16 + (lane & 15);
    const float* ar = gemb + row * HDIM;

    v8f acc = {};
    #pragma unroll
    for (int kt = 0; kt < 8; ++kt) {
        v16bf a  = load_a_bf16(ar, kt, khalf);
        v16bf bm = *(const v16bf*)&ldsB[(((kt << 3) | wave) << 9) + (lane << 4)];
        acc = __builtin_amdgcn_wmma_f32_16x16x32_bf16(
            false, a, false, bm, (short)0, acc, false, false);
    }

    const int   c   = ntl * 16 + (lane & 15);
    const float bpv = bp[c];
    #pragma unroll
    for (int r = 0; r < 8; ++r) {
        int m = mt * 16 + r + khalf * 8;
        ctx[m * HDIM + c] = acc[r] + bpv;
    }
}

// -------- kernel 5: out[n] = context[batch[n]]  (~205 MB stream) ----------
__global__ __launch_bounds__(256)
void gci_scatter_kernel(const float* __restrict__ ctx,
                        const int* __restrict__ batch,
                        float* __restrict__ out, int n) {
    int id = blockIdx.x * 256 + threadIdx.x;
    if (id >= n * 64) return;
    int node = id >> 6;          // 64 float4 per 256-wide row
    int q    = id & 63;
    int g    = batch[node];
    f4 v = *(const f4*)(ctx + g * HDIM + q * 4);
    __builtin_nontemporal_store(v, (f4*)(out + (long)node * HDIM + q * 4));
}

// -------- launcher -------------------------------------------------------
extern "C" void kernel_launch(void* const* d_in, const int* in_sizes, int n_in,
                              void* d_out, int out_size, void* d_ws, size_t ws_size,
                              hipStream_t stream) {
    const float* x     = (const float*)d_in[0];
    const int*   batch = (const int*)d_in[1];   // harness: integer -> int32
    const float* W1    = (const float*)d_in[2];
    const float* b1    = (const float*)d_in[3];
    const float* W2    = (const float*)d_in[4];
    const float* b2    = (const float*)d_in[5];
    const float* Wp    = (const float*)d_in[6];
    const float* bp    = (const float*)d_in[7];
    float*       out   = (float*)d_out;

    const int n = in_sizes[1];                  // number of nodes (200000)

    // workspace carve-up (all 256B-aligned slices): s[N], w[N], gemb, ctx
    const int NPAD = 200704;                    // 128-multiple >= N
    float* ws    = (float*)d_ws;
    float* s     = ws;
    float* wnode = s + NPAD;
    float* gemb  = wnode + NPAD;
    float* ctx   = gemb + GSEG * HDIM;

    gci_score_kernel<<<(n + 127) / 128, 256, 0, stream>>>(x, W1, b1, W2, b2, s, n);
    gci_segsoft_kernel<<<GSEG, 256, 0, stream>>>(batch, s, wnode, n);
    gci_pool_kernel<<<GSEG, 256, 0, stream>>>(x, batch, wnode, gemb, n);
    gci_proj_kernel<<<64, 256, 0, stream>>>(gemb, Wp, bp, ctx);
    gci_scatter_kernel<<<(n * 64 + 255) / 256, 256, 0, stream>>>(ctx, batch, out, n);
}
